// GCNLayer_7430293422435
// MI455X (gfx1250) — matrix operand order repaired and checked by reading, not compiled, not run on hardware
//
#include <hip/hip_runtime.h>
#include <hip/hip_bf16.h>

typedef __attribute__((ext_vector_type(16))) __bf16          v16bf;
typedef __attribute__((ext_vector_type(8)))  float           v8f;
typedef __attribute__((ext_vector_type(16))) unsigned short  v16u;
typedef __attribute__((ext_vector_type(8)))  unsigned int    v8u;

#define IN_F   512
#define OUT_F  512
#define M_BLK  128
#define N_BLK  64
#define KC     32            // K per WMMA
#define NKC    (IN_F / KC)   // 16 K-chunks

__device__ __forceinline__ unsigned short f32_to_bf16_rne(float f) {
    unsigned int u = __float_as_uint(f);
    unsigned int r = u + 0x7FFFu + ((u >> 16) & 1u);
    return (unsigned short)(r >> 16);
}

__device__ __forceinline__ unsigned int pack2_bf16(float x, float y) {
    return (unsigned int)f32_to_bf16_rne(x) | ((unsigned int)f32_to_bf16_rne(y) << 16);
}

// ---------------- one-shot f32 -> bf16 conversion (A and W) ----------------
__global__ void cvt_f32_to_bf16(const float* __restrict__ src,
                                unsigned int* __restrict__ dst, size_t nQuads)
{
    size_t i = (size_t)blockIdx.x * blockDim.x + threadIdx.x;
    size_t stride = (size_t)gridDim.x * blockDim.x;
    for (; i < nQuads; i += stride) {
        float4 f = ((const float4*)src)[i];
        uint2 o;
        o.x = pack2_bf16(f.x, f.y);
        o.y = pack2_bf16(f.z, f.w);
        ((uint2*)dst)[i] = o;
    }
}

// ---------------- GEMM: support = A(MxK) * W(KxN), bf16 WMMA, f32 accum ----------------
__global__ __launch_bounds__(256) void gcn_gemm_wmma(
    const unsigned short* __restrict__ Abf,   // M x IN_F bf16 (row major)
    const unsigned short* __restrict__ Wbf,   // IN_F x OUT_F bf16 (row major)
    float* __restrict__ C, int M)
{
    // packed bf16 B slice: [kc][nb][lane][elem]; each (kc,nb) fragment = 32 lanes x 16 halfs
    __shared__ unsigned short bs[NKC * 4 * 32 * 16];   // 64 KB

    const int tid = threadIdx.x;
    const int n0  = blockIdx.x * N_BLK;
    const int m0  = blockIdx.y * M_BLK;

    // pure-permute LDS fill from preconverted bf16 weights
    for (int e = tid; e < IN_F * N_BLK; e += 256) {
        int k  = e >> 6;
        int n  = e & 63;
        int kc = k >> 5, kk = k & 31;
        int nb = n >> 4, nl = n & 15;
        int lane = nl + ((kk & 8) << 1);
        int elem = (kk & 7) | ((kk & 16) >> 1);
        bs[(((kc << 2) + nb) * 32 + lane) * 16 + elem] = Wbf[(size_t)k * OUT_F + n0 + n];
    }
    __syncthreads();

    const int wave  = tid >> 5;
    const int lane  = tid & 31;
    const int hi    = lane >> 4;
    const int mBase = m0 + wave * 16;

    int mRow = mBase + (lane & 15);
    if (mRow > M - 1) mRow = M - 1;            // clamp tail loads; stores guarded
    const unsigned short* aRow = Abf + (size_t)mRow * IN_F + (hi ? 8 : 0);

    v8f acc0 = {}, acc1 = {}, acc2 = {}, acc3 = {};

#pragma unroll
    for (int kc = 0; kc < NKC; ++kc) {
        // A fragment: 16 bf16 per lane: K = c..c+7 and c+16..c+23  (c = kc*32 + hi*8)
        const uint4* ap = (const uint4*)(aRow + kc * KC);
        uint4 alo = ap[0];         // 8 bf16, K = c..c+7
        uint4 ahi = ap[2];         // 8 bf16, K = c+16..c+23
        v8u au;
        au[0] = alo.x; au[1] = alo.y; au[2] = alo.z; au[3] = alo.w;
        au[4] = ahi.x; au[5] = ahi.y; au[6] = ahi.z; au[7] = ahi.w;
        v16bf af = __builtin_bit_cast(v16bf, au);

        const unsigned short* bbase = bs + ((size_t)(kc << 2) * 32 + lane) * 16;
        v16bf b0 = __builtin_bit_cast(v16bf, *(const v16u*)(bbase + 0 * 512));
        v16bf b1 = __builtin_bit_cast(v16bf, *(const v16u*)(bbase + 1 * 512));
        v16bf b2 = __builtin_bit_cast(v16bf, *(const v16u*)(bbase + 2 * 512));
        v16bf b3 = __builtin_bit_cast(v16bf, *(const v16u*)(bbase + 3 * 512));

        acc0 = __builtin_amdgcn_wmma_f32_16x16x32_bf16(false, af, false, b0, (short)0, acc0, false, false);
        acc1 = __builtin_amdgcn_wmma_f32_16x16x32_bf16(false, af, false, b1, (short)0, acc1, false, false);
        acc2 = __builtin_amdgcn_wmma_f32_16x16x32_bf16(false, af, false, b2, (short)0, acc2, false, false);
        acc3 = __builtin_amdgcn_wmma_f32_16x16x32_bf16(false, af, false, b3, (short)0, acc3, false, false);
    }

    const int col   = n0 + (lane & 15);
    const int rBase = mBase + (hi ? 8 : 0);
#pragma unroll
    for (int v = 0; v < 8; ++v) {
        int r = rBase + v;
        if (r < M) {
            float* cp = C + (size_t)r * OUT_F + col;
            cp[0]  = acc0[v];
            cp[16] = acc1[v];
            cp[32] = acc2[v];
            cp[48] = acc3[v];
        }
    }
}

// ---------------- CSR build: histogram -> exclusive scan -> scatter ----------------
__global__ void zero_i32(int* __restrict__ p, int n) {
    int i = blockIdx.x * blockDim.x + threadIdx.x;
    if (i < n) p[i] = 0;
}

__global__ void hist_rows(const int* __restrict__ erow, int* __restrict__ counts, int E) {
    int i = blockIdx.x * blockDim.x + threadIdx.x;
    int stride = gridDim.x * blockDim.x;
    for (; i < E; i += stride) atomicAdd(&counts[erow[i]], 1);
}

__global__ __launch_bounds__(256) void scan_block_sums(
    const int* __restrict__ counts, int* __restrict__ partials, int M)
{
    __shared__ int s[256];
    int i = blockIdx.x * 256 + threadIdx.x;
    s[threadIdx.x] = (i < M) ? counts[i] : 0;
    __syncthreads();
    for (int off = 128; off > 0; off >>= 1) {
        if (threadIdx.x < off) s[threadIdx.x] += s[threadIdx.x + off];
        __syncthreads();
    }
    if (threadIdx.x == 0) partials[blockIdx.x] = s[0];
}

__global__ __launch_bounds__(512) void scan_top(
    int* __restrict__ partials, int nb, int* __restrict__ totalOut)
{
    __shared__ int s[512];
    int t = threadIdx.x;
    int v = (t < nb) ? partials[t] : 0;
    s[t] = v;
    __syncthreads();
    for (int off = 1; off < 512; off <<= 1) {
        int add = (t >= off) ? s[t - off] : 0;
        __syncthreads();
        s[t] += add;
        __syncthreads();
    }
    if (t < nb) partials[t] = s[t] - v;      // exclusive
    if (t == 511) *totalOut = s[511];        // rowptr[M] = E
}

__global__ __launch_bounds__(256) void scan_chunks(
    const int* __restrict__ counts, const int* __restrict__ partials,
    int* __restrict__ rowptr, int* __restrict__ cursor, int M)
{
    __shared__ int s[256];
    int i = blockIdx.x * 256 + threadIdx.x;
    int v = (i < M) ? counts[i] : 0;
    s[threadIdx.x] = v;
    __syncthreads();
    for (int off = 1; off < 256; off <<= 1) {
        int add = (threadIdx.x >= off) ? s[threadIdx.x - off] : 0;
        __syncthreads();
        s[threadIdx.x] += add;
        __syncthreads();
    }
    if (i < M) {
        int excl = s[threadIdx.x] - v + partials[blockIdx.x];
        rowptr[i] = excl;
        cursor[i] = excl;
    }
}

__global__ void csr_scatter(const int* __restrict__ erow, const int* __restrict__ ecol,
                            const float* __restrict__ eval, int* __restrict__ cursor,
                            int* __restrict__ csr_col, float* __restrict__ csr_val, int E)
{
    int i = blockIdx.x * blockDim.x + threadIdx.x;
    int stride = gridDim.x * blockDim.x;
    for (; i < E; i += stride) {
        int r = erow[i];
        int pos = atomicAdd(&cursor[r], 1);
        csr_col[pos] = ecol[i];
        csr_val[pos] = eval[i];
    }
}

// ---------------- Row-gather SpMM + fused ReLU: no data-path atomics ----------------
__global__ __launch_bounds__(128) void spmm_row_relu(
    const int* __restrict__ rowptr, const int* __restrict__ csr_col,
    const float* __restrict__ csr_val, const float* __restrict__ sup,
    float* __restrict__ out)
{
    const int r   = blockIdx.x;
    const int beg = rowptr[r];
    const int end = rowptr[r + 1];
    const int t4  = threadIdx.x * 4;

    float4 acc = make_float4(0.f, 0.f, 0.f, 0.f);
    int j = beg;
    for (; j + 1 < end; j += 2) {
        int   c0 = csr_col[j],   c1 = csr_col[j + 1];   // block-uniform -> scalar loads
        float v0 = csr_val[j],   v1 = csr_val[j + 1];
        float4 x0 = *(const float4*)(sup + (size_t)c0 * OUT_F + t4);
        float4 x1 = *(const float4*)(sup + (size_t)c1 * OUT_F + t4);
        acc.x += v0 * x0.x + v1 * x1.x;
        acc.y += v0 * x0.y + v1 * x1.y;
        acc.z += v0 * x0.z + v1 * x1.z;
        acc.w += v0 * x0.w + v1 * x1.w;
    }
    if (j < end) {
        int   c = csr_col[j];
        float v = csr_val[j];
        float4 x = *(const float4*)(sup + (size_t)c * OUT_F + t4);
        acc.x += v * x.x; acc.y += v * x.y; acc.z += v * x.z; acc.w += v * x.w;
    }
    acc.x = fmaxf(acc.x, 0.f);
    acc.y = fmaxf(acc.y, 0.f);
    acc.z = fmaxf(acc.z, 0.f);
    acc.w = fmaxf(acc.w, 0.f);
    *(float4*)(out + (size_t)r * OUT_F + t4) = acc;
}

extern "C" void kernel_launch(void* const* d_in, const int* in_sizes, int n_in,
                              void* d_out, int out_size, void* d_ws, size_t ws_size,
                              hipStream_t stream) {
    const float* features = (const float*)d_in[0];
    const float* weight   = (const float*)d_in[1];
    const int*   erow     = (const int*)d_in[2];
    const int*   ecol     = (const int*)d_in[3];
    const float* eval     = (const float*)d_in[4];
    float* out = (float*)d_out;

    const int M = in_sizes[0] / IN_F;   // 100000
    const int E = in_sizes[2];          // 3200000
    const int NB = (M + 255) / 256;     // scan chunks

    // ws layout
    float* support  = (float*)d_ws;                        // M * OUT_F floats
    int*   rowptr   = (int*)(support + (size_t)M * OUT_F); // M+1
    int*   cursor   = rowptr + (M + 1);                    // M
    int*   csr_col  = cursor + M;                          // E
    float* csr_val  = (float*)(csr_col + E);               // E
    int*   partials = (int*)(csr_val + E);                 // NB
    uintptr_t pa = (uintptr_t)(partials + NB);
    pa = (pa + 255) & ~(uintptr_t)255;
    unsigned short* abf = (unsigned short*)pa;             // M * IN_F bf16
    unsigned short* wbf = abf + (size_t)M * IN_F;          // IN_F * OUT_F bf16 (offset is 256B-aligned)

    // --- one-shot bf16 conversion of A and W ---
    cvt_f32_to_bf16<<<2048, 256, 0, stream>>>(features, (unsigned int*)abf,
                                              ((size_t)M * IN_F) / 4);
    cvt_f32_to_bf16<<<256, 256, 0, stream>>>(weight, (unsigned int*)wbf,
                                             ((size_t)IN_F * OUT_F) / 4);

    // --- dense GEMM via WMMA ---
    dim3 grid(OUT_F / N_BLK, (M + M_BLK - 1) / M_BLK);
    gcn_gemm_wmma<<<grid, 256, 0, stream>>>(abf, wbf, support, M);

    // --- COO -> CSR ---
    zero_i32<<<NB, 256, 0, stream>>>(cursor, M);
    hist_rows<<<2048, 256, 0, stream>>>(erow, cursor, E);
    scan_block_sums<<<NB, 256, 0, stream>>>(cursor, partials, M);
    scan_top<<<1, 512, 0, stream>>>(partials, NB, rowptr + M);
    scan_chunks<<<NB, 256, 0, stream>>>(cursor, partials, rowptr, cursor, M);
    csr_scatter<<<2048, 256, 0, stream>>>(erow, ecol, eval, cursor, csr_col, csr_val, E);

    // --- row-gather SpMM + fused ReLU ---
    spmm_row_relu<<<M, 128, 0, stream>>>(rowptr, csr_col, csr_val, support, out);
}
